// FrequencyDomainAttention_56573309223495
// MI455X (gfx1250) — compile-verified, matrix operand order checked by reading, not run on hardware
//
#include <hip/hip_runtime.h>

typedef __attribute__((ext_vector_type(16))) __bf16 v16bf;
typedef __attribute__((ext_vector_type(8)))  float  v8f;

#define DEVFN __device__ __forceinline__

// Async global->LDS path (GLOBAL_LOAD_ASYNC_TO_LDS_B128 + s_wait_asynccnt),
// guarded so the fallback compiles identically on toolchains without it.
#if defined(__has_builtin)
#if __has_builtin(__builtin_amdgcn_global_load_async_to_lds_b128) && \
    __has_builtin(__builtin_amdgcn_s_wait_asynccnt)
#define FDA_ASYNC 1
#endif
#endif
#ifndef FDA_ASYNC
#define FDA_ASYNC 0
#endif

// Pointer types matching the builtin prototype "vV4i*1V4i*3IiIi":
// (global int4 vector*, LDS int4 vector*, imm offset, imm cpol)
typedef int v4i_t __attribute__((vector_size(4 * sizeof(int))));
typedef __attribute__((address_space(1))) v4i_t* as1_v4i_p;
typedef __attribute__((address_space(3))) v4i_t* as3_v4i_p;

namespace {

constexpr int kB  = 2;
constexpr int kN  = 2048;
constexpr int kC  = 1024;
constexpr int kH  = 16;
constexpr int kDh = 64;
constexpr int kFv = 1025;          // valid rfft bins
constexpr int kFp = 1152;          // padded to 9*128 -> every GEMM dim tiles exactly
constexpr int kFT = kFp / 16;      // 72 = 9*8 sixteen-tiles
constexpr float kInvSqrtN = 0.022097086912079608f;   // 1/sqrt(2048)
constexpr float kScale    = 0.125f;                  // Dh^-0.5

constexpr int BM = 128, BK = 32;
constexpr int LSTR = 40;           // LDS row stride (ushorts): 32 + 8 pad, keeps 16B align

union Frag {
  v16bf v;
  unsigned u[8];
  uint4 u4[2];
};

DEVFN unsigned short f2bf(float f) {
  unsigned u = __float_as_uint(f);
  u += 0x7FFFu + ((u >> 16) & 1u);   // round-to-nearest-even
  return (unsigned short)(u >> 16);
}

DEVFN v8f wmma_bf16(v16bf a, v16bf b, v8f c) {
  return __builtin_amdgcn_wmma_f32_16x16x32_bf16(false, a, false, b, (short)0, c,
                                                 false, false);
}

// 16B global -> LDS copy; async on toolchains that expose the gfx1250 builtin.
DEVFN void g2lds16(const unsigned short* src, unsigned short* dst) {
#if FDA_ASYNC
  __builtin_amdgcn_global_load_async_to_lds_b128((as1_v4i_p)src, (as3_v4i_p)dst, 0, 0);
#else
  *(uint4*)dst = *(const uint4*)src;
#endif
}

DEVFN void g2lds_wait() {
#if FDA_ASYNC
  __builtin_amdgcn_s_wait_asynccnt(0);
#endif
}

// ---------------------------------------------------------------------------
// Fill bf16 DFT (2Fp x N) and iDFT (N x 2Fp) matrices (ortho norm).
// DFT rows [0,Fp): cos(2*pi*f*n/N)/sqrt(N); rows [Fp,2Fp): -sin(...)/sqrt(N).
// iDFT cols [0,Fp): c_f*cos/sqrt(N); cols [Fp,2Fp): -c_f*sin/sqrt(N),
// c_f = 1 for f==0 or f==N/2 else 2.  All entries with f>=1025 are zero, so
// the frequency padding is exactly zero end-to-end.
// ---------------------------------------------------------------------------
__global__ void fda_gen_dft(unsigned short* __restrict__ dft,
                            unsigned short* __restrict__ idft) {
  const long total = (long)2 * kFp * kN;
  long idx = (long)blockIdx.x * blockDim.x + threadIdx.x;
  if (idx >= total) return;

  {   // forward DFT element (r, n)
    int r = (int)(idx / kN), n = (int)(idx % kN);
    int f = (r < kFp) ? r : r - kFp;
    float v = 0.f;
    if (f < kFv) {
      int m = (f * n) & (kN - 1);
      float ph = (float)m * (6.283185307179586f / (float)kN);
      v = ((r < kFp) ? __cosf(ph) : -__sinf(ph)) * kInvSqrtN;
    }
    dft[idx] = f2bf(v);
  }
  {   // inverse DFT element (n2, j)
    int n2 = (int)(idx / (2 * kFp)), j = (int)(idx % (2 * kFp));
    int f = (j < kFp) ? j : j - kFp;
    float v = 0.f;
    if (f < kFv) {
      float cf = (f == 0 || f == kN / 2) ? 1.f : 2.f;
      int m = (f * n2) & (kN - 1);
      float ph = (float)m * (6.283185307179586f / (float)kN);
      v = ((j < kFp) ? __cosf(ph) : -__sinf(ph)) * cf * kInvSqrtN;
    }
    idft[idx] = f2bf(v);
  }
}

__global__ void fda_cvt_bf16(const float* __restrict__ src,
                             unsigned short* __restrict__ dst, long n) {
  long i = (long)blockIdx.x * blockDim.x + threadIdx.x;
  if (i < n) dst[i] = f2bf(src[i]);
}

// ---------------------------------------------------------------------------
// Batched bf16 NN GEMM, fully tiled (no bounds guards), LDS double-buffered.
// A tiles are staged with async global->LDS copies (ASYNCcnt); B tiles are
// register-staged because they are transposed into LDS.
// C = A(MxK) @ B(KxN) [+ bias].  Batch z = ((b*cntH + h)*cntP + p), each
// operand gets (b,h,p) strides.  256 threads = 8 waves, 4(M) x 2(N) wave
// grid; wave tile 32 x (BN_/2).  BN_ in {128, 64}.
// ---------------------------------------------------------------------------
template <int BN_>
__global__ __launch_bounds__(256) void fda_gemm_bf16(
    const unsigned short* __restrict__ A,
    const unsigned short* __restrict__ B,
    void* __restrict__ C,
    const float* __restrict__ bias,
    int M, int Nc, int K, int lda, int ldb, int ldc,
    long offA0, long sAb, long sAh, long sAp,
    long offB0, long sBb, long sBh, long sBp,
    long offC0, long sCb, long sCh, long sCp,
    int cntH, int cntP, int outF32) {
  constexpr int TN = BN_ / 32;          // WMMA tiles per wave in N
  constexpr int BV = BN_ / 64;          // B-tile vec8 loads per thread
  constexpr int BKSTEP = 2048 / BN_;    // B-tile k-rows covered per 256 vecs

  __shared__ alignas(16) unsigned short sA[2][BM * LSTR];
  __shared__ alignas(16) unsigned short sBt[2][BN_ * LSTR];

  const int tid = threadIdx.x;
  const int z = blockIdx.z;
  const int p = z % cntP;
  const int h = (z / cntP) % cntH;
  const int b = z / (cntP * cntH);
  A += offA0 + (long)b * sAb + (long)h * sAh + (long)p * sAp;
  B += offB0 + (long)b * sBb + (long)h * sBh + (long)p * sBp;
  const long coff = offC0 + (long)b * sCb + (long)h * sCh + (long)p * sCp;

  const int rowBase = blockIdx.y * BM;
  const int colBase = blockIdx.x * BN_;

  const int lane = tid & 31;
  const int wave = tid >> 5;
  const int lmod = lane & 15, lhalf = lane >> 4;
  const int wm = wave & 3;              // 4 M-groups of 32 rows
  const int wn = wave >> 2;             // 2 N-groups of BN_/2 cols

  // fixed per-thread staging coordinates
  const int arow = tid >> 2;            // A tile: 128 rows x 4 vec8/row
  const int acol = (tid & 3) * 8;
  const int brow = tid / (BN_ / 8);     // B tile: BK rows x BN_/8 vec8/row
  const int bcol = (tid % (BN_ / 8)) * 8;

  union U { uint4 v; unsigned short s[8]; };
  U rb[BV];

  auto asyncA = [&](int k0, int buf) {
#pragma unroll
    for (int i = 0; i < 2; ++i)
      g2lds16(A + (long)(rowBase + arow + i * 64) * lda + k0 + acol,
              &sA[buf][(arow + i * 64) * LSTR + acol]);
  };
  auto loadB = [&](int k0) {
#pragma unroll
    for (int i = 0; i < BV; ++i)
      rb[i].v = *(const uint4*)(B + (long)(k0 + brow + i * BKSTEP) * ldb + colBase + bcol);
  };
  auto stageB = [&](int buf) {
#pragma unroll
    for (int i = 0; i < BV; ++i)
#pragma unroll
      for (int e = 0; e < 8; ++e)
        sBt[buf][(bcol + e) * LSTR + (brow + i * BKSTEP)] = rb[i].s[e];
  };

  asyncA(0, 0);
  loadB(0);
  stageB(0);
  g2lds_wait();
  __syncthreads();

  v8f acc[2][TN] = {};
  int cur = 0;
  for (int k0 = 0; k0 < K; k0 += BK) {
    const bool notLast = (k0 + BK) < K;
    if (notLast) {
      asyncA(k0 + BK, cur ^ 1);                     // async copy next A tile
      loadB(k0 + BK);                               // overlap global with WMMA
    }
    if (k0 + 2 * BK < K) {                          // speculative prefetch k+2
      __builtin_prefetch(A + (long)(rowBase + arow) * lda + k0 + 2 * BK + acol, 0, 1);
      __builtin_prefetch(B + (long)(k0 + 2 * BK + brow) * ldb + colBase + bcol, 0, 1);
    }

    Frag fa[2], fb[TN];
#pragma unroll
    for (int i = 0; i < 2; ++i) {
      const unsigned short* pa = &sA[cur][(wm * 32 + i * 16 + lmod) * LSTR + lhalf * 8];
      fa[i].u4[0] = *(const uint4*)pa;
      fa[i].u4[1] = *(const uint4*)(pa + 16);
    }
#pragma unroll
    for (int j = 0; j < TN; ++j) {
      const unsigned short* pb =
          &sBt[cur][(wn * (BN_ / 2) + j * 16 + lmod) * LSTR + lhalf * 8];
      fb[j].u4[0] = *(const uint4*)pb;
      fb[j].u4[1] = *(const uint4*)(pb + 16);
    }
#pragma unroll
    for (int i = 0; i < 2; ++i)
#pragma unroll
      for (int j = 0; j < TN; ++j)
        acc[i][j] = wmma_bf16(fa[i].v, fb[j].v, acc[i][j]);

    if (notLast) stageB(cur ^ 1);
    g2lds_wait();                                   // async A writes visible
    __syncthreads();                                // one barrier per K-step
    cur ^= 1;
  }

  // epilogue: C layout lane -> N, VGPR v -> row v + 8*(lane/16)
#pragma unroll
  for (int j = 0; j < TN; ++j) {
    const int colg = colBase + wn * (BN_ / 2) + j * 16 + lmod;
    const float bj = (bias != nullptr) ? bias[colg] : 0.f;
#pragma unroll
    for (int i = 0; i < 2; ++i) {
#pragma unroll
      for (int v = 0; v < 8; ++v) {
        const long idx =
            coff + (long)(rowBase + wm * 32 + i * 16 + v + 8 * lhalf) * ldc + colg;
        const float val = acc[i][j][v] + bj;
        if (outF32) ((float*)C)[idx] = val;
        else ((unsigned short*)C)[idx] = f2bf(val);
      }
    }
  }
}

// ---------------------------------------------------------------------------
// scores[b,h,f,g] = scale * (qr[f,:]·kr[g,:] - qi[f,:]·ki[g,:])
// NT GEMM, K = Dh = 64 = two 16x16x32 WMMA steps per complex part; fragments
// loaded straight from global (rows contiguous for both A and B^T operands).
// qkv layout: (b, part, Fp, 3C); q at col h*64, k at col C + h*64.
// Grid is exact: 72 g-tiles x 9 blocks of 8 f-tile waves x (B*H).
// ---------------------------------------------------------------------------
__global__ __launch_bounds__(256) void fda_scores(
    const unsigned short* __restrict__ qkv, float* __restrict__ scores) {
  const int lane = threadIdx.x & 31;
  const int wave = threadIdx.x >> 5;
  const int gT = blockIdx.x;                  // 0..71
  const int fT = blockIdx.y * 8 + wave;       // 0..71, wave-uniform
  const int bh = blockIdx.z;
  const int b = bh / kH, h = bh % kH;

  const int lmod = lane & 15, lhalf = lane >> 4;
  const int ld = 3 * kC;
  const long qrB = (long)(b * 2 + 0) * kFp * ld + (long)h * kDh;
  const long qiB = (long)(b * 2 + 1) * kFp * ld + (long)h * kDh;
  const int fr = fT * 16 + lmod;
  const int gr = gT * 16 + lmod;

  v8f acc = {};
#pragma unroll
  for (int part = 0; part < 2; ++part) {
    const long qBase = (part ? qiB : qrB);
    const long kBase = qBase + kC;
#pragma unroll
    for (int s = 0; s < 2; ++s) {
      Frag fq, fk;
      const unsigned short* pq = qkv + qBase + (long)fr * ld + s * 32 + lhalf * 8;
      fq.u4[0] = *(const uint4*)pq;
      fq.u4[1] = *(const uint4*)(pq + 16);
      const unsigned short* pk = qkv + kBase + (long)gr * ld + s * 32 + lhalf * 8;
      fk.u4[0] = *(const uint4*)pk;
      fk.u4[1] = *(const uint4*)(pk + 16);
      if (part) {                 // negate qi: flip bf16 sign bits
#pragma unroll
        for (int e = 0; e < 8; ++e) fq.u[e] ^= 0x80008000u;
      }
      acc = wmma_bf16(fq.v, fk.v, acc);
    }
  }

  const long sBase = ((long)bh * kFp + fT * 16) * kFp + gT * 16;
#pragma unroll
  for (int v = 0; v < 8; ++v) {
    const int row = v + 8 * lhalf;
    scores[sBase + (long)row * kFp + lmod] = acc[v] * kScale;
  }
}

// ---------------------------------------------------------------------------
// Row softmax over g in [0,1025); writes bf16 attn with cols [1025,Fp)=0.
// One 256-thread block per row (B*H*Fp rows).
// ---------------------------------------------------------------------------
__global__ __launch_bounds__(256) void fda_softmax(
    const float* __restrict__ scores, unsigned short* __restrict__ attn) {
  const long row = blockIdx.x;
  const float* s = scores + row * kFp;
  unsigned short* a = attn + row * kFp;
  __shared__ float red[256];
  const int tid = threadIdx.x;

  float mx = -1e30f;
  for (int g = tid; g < kFv; g += 256) mx = fmaxf(mx, s[g]);
  red[tid] = mx;
  __syncthreads();
  for (int st = 128; st > 0; st >>= 1) {
    if (tid < st) red[tid] = fmaxf(red[tid], red[tid + st]);
    __syncthreads();
  }
  mx = red[0];
  __syncthreads();

  float sum = 0.f;
  for (int g = tid; g < kFv; g += 256) sum += __expf(s[g] - mx);
  red[tid] = sum;
  __syncthreads();
  for (int st = 128; st > 0; st >>= 1) {
    if (tid < st) red[tid] += red[tid + st];
    __syncthreads();
  }
  const float inv = 1.f / red[0];

  for (int g = tid; g < kFp; g += 256) {
    const float v = (g < kFv) ? __expf(s[g] - mx) * inv : 0.f;
    a[g] = f2bf(v);
  }
}

}  // namespace

// ---------------------------------------------------------------------------
extern "C" void kernel_launch(void* const* d_in, const int* in_sizes, int n_in,
                              void* d_out, int out_size, void* d_ws, size_t ws_size,
                              hipStream_t stream) {
  (void)in_sizes; (void)n_in; (void)out_size; (void)ws_size;
  const float* x     = (const float*)d_in[0];
  const float* Wqkv  = (const float*)d_in[1];
  const float* bqkv  = (const float*)d_in[2];
  const float* Wproj = (const float*)d_in[3];
  const float* bproj = (const float*)d_in[4];
  float* out = (float*)d_out;

  char* w = (char*)d_ws;
  auto take = [&](size_t bytes) -> char* {
    char* p = w;
    w += (bytes + 255) & ~(size_t)255;
    return p;
  };
  unsigned short* dft    = (unsigned short*)take((size_t)2 * kFp * kN * 2);     // (2Fp, N)
  unsigned short* idft   = (unsigned short*)take((size_t)kN * 2 * kFp * 2);     // (N, 2Fp)
  unsigned short* xb     = (unsigned short*)take((size_t)kB * kN * kC * 2);     // (B, N, C)
  unsigned short* wqkvb  = (unsigned short*)take((size_t)kC * 3 * kC * 2);      // (C, 3C)
  unsigned short* wprojb = (unsigned short*)take((size_t)kC * kC * 2);          // (C, C)
  unsigned short* xfb    = (unsigned short*)take((size_t)kB * 2 * kFp * kC * 2);// (B, 2Fp, C)
  unsigned short* qkvb   = (unsigned short*)take((size_t)kB * 2 * kFp * 3 * kC * 2); // (B,2,Fp,3C)
  float*          scoresb= (float*)take((size_t)kB * kH * kFp * kFp * 4);       // (B,H,Fp,Fp)
  unsigned short* attnb  = (unsigned short*)take((size_t)kB * kH * kFp * kFp * 2);
  unsigned short* outb   = (unsigned short*)take((size_t)kB * 2 * kFp * kC * 2);// (B, 2Fp, C)
  unsigned short* timeb  = (unsigned short*)take((size_t)kB * kN * kC * 2);     // (B, N, C)

  const int TPB = 256;

  // 0) DFT / iDFT matrices + bf16 conversions
  fda_gen_dft<<<dim3((2 * kFp * kN + TPB - 1) / TPB), TPB, 0, stream>>>(dft, idft);
  fda_cvt_bf16<<<dim3(((long)kB * kN * kC + TPB - 1) / TPB), TPB, 0, stream>>>(x, xb, (long)kB * kN * kC);
  fda_cvt_bf16<<<dim3(((long)kC * 3 * kC + TPB - 1) / TPB), TPB, 0, stream>>>(Wqkv, wqkvb, (long)kC * 3 * kC);
  fda_cvt_bf16<<<dim3(((long)kC * kC + TPB - 1) / TPB), TPB, 0, stream>>>(Wproj, wprojb, (long)kC * kC);

  // 1) xf = DFT @ x : per batch, M=2Fp=2304, K=N=2048, N=C -> xfb (bf16)
  fda_gemm_bf16<128><<<dim3(kC / 128, (2 * kFp) / 128, kB), TPB, 0, stream>>>(
      dft, xb, xfb, nullptr,
      2 * kFp, kC, kN, kN, kC, kC,
      0, 0, 0, 0,
      0, (long)kN * kC, 0, 0,
      0, (long)2 * kFp * kC, 0, 0,
      1, 1, 0);

  // 2) qkv = xf @ Wqkv (+bias on real part only) : M=Fp, K=C, N=3C
  fda_gemm_bf16<128><<<dim3(3 * kC / 128, kFp / 128, kB), TPB, 0, stream>>>(
      xfb, wqkvb, qkvb, bqkv,                           // real part, with bias
      kFp, 3 * kC, kC, kC, 3 * kC, 3 * kC,
      0, (long)2 * kFp * kC, 0, 0,
      0, 0, 0, 0,
      0, (long)2 * kFp * 3 * kC, 0, 0,
      1, 1, 0);
  fda_gemm_bf16<128><<<dim3(3 * kC / 128, kFp / 128, kB), TPB, 0, stream>>>(
      xfb, wqkvb, qkvb, nullptr,                        // imag part, no bias
      kFp, 3 * kC, kC, kC, 3 * kC, 3 * kC,
      (long)kFp * kC, (long)2 * kFp * kC, 0, 0,
      0, 0, 0, 0,
      (long)kFp * 3 * kC, (long)2 * kFp * 3 * kC, 0, 0,
      1, 1, 0);

  // 3) scores.real = scale*(qr@kr^T - qi@ki^T)
  fda_scores<<<dim3(kFT, kFT / 8, kB * kH), TPB, 0, stream>>>(qkvb, scoresb);

  // 4) softmax over valid freqs -> bf16 attn (padded cols zeroed)
  fda_softmax<<<dim3(kB * kH * kFp), TPB, 0, stream>>>(scoresb, attnb);

  // 5) out_{r,i} = attn @ v_{r,i} : batch (b,h,part), M=Fp, K=Fp, N=Dh=64
  fda_gemm_bf16<64><<<dim3(1, kFp / 128, kB * kH * 2), TPB, 0, stream>>>(
      attnb, qkvb, outb, nullptr,
      kFp, kDh, kFp, kFp, 3 * kC, kC,
      0, (long)kH * kFp * kFp, (long)kFp * kFp, 0,
      (long)2 * kC, (long)2 * kFp * 3 * kC, (long)kDh, (long)kFp * 3 * kC,
      0, (long)2 * kFp * kC, (long)kDh, (long)kFp * kC,
      kH, 2, 0);

  // 6) time = iDFT @ [out_r; out_i] : M=N, K=2Fp (real+imag fused), N=C
  fda_gemm_bf16<128><<<dim3(kC / 128, kN / 128, kB), TPB, 0, stream>>>(
      idft, outb, timeb, nullptr,
      kN, kC, 2 * kFp, 2 * kFp, kC, kC,
      0, 0, 0, 0,
      0, (long)2 * kFp * kC, 0, 0,
      0, (long)kN * kC, 0, 0,
      1, 1, 0);

  // 7) result = time @ Wproj + bproj : fp32 out
  fda_gemm_bf16<128><<<dim3(kC / 128, kN / 128, kB), TPB, 0, stream>>>(
      timeb, wprojb, out, bproj,
      kN, kC, kC, kC, kC, kC,
      0, (long)kN * kC, 0, 0,
      0, 0, 0, 0,
      0, (long)kN * kC, 0, 0,
      1, 1, 1);
}
